// BinaryMLP_66640712564850
// MI455X (gfx1250) — compile-verified, matrix operand order checked by reading, not run on hardware
//
#include <hip/hip_runtime.h>

#define D_MODEL 2048
#define HIDDEN  5461
#define H_PAD   5504   // 86 * 64
#define M_TOK   16384  // 4 * 4096

typedef __attribute__((ext_vector_type(16))) __bf16 v16bf;
typedef __attribute__((ext_vector_type(8)))  __bf16 v8bf;
typedef __attribute__((ext_vector_type(8)))  float  v8f;

union FragBF { v16bf v; v8bf h[2]; };

#if __has_builtin(__builtin_amdgcn_global_load_async_to_lds_b128)
#define HAS_ASYNC_LDS 1
#else
#define HAS_ASYNC_LDS 0
#endif

typedef int v4i_vs __attribute__((vector_size(16)));
typedef __attribute__((address_space(1))) v4i_vs* gv4p;  // global (AS1) int4*
typedef __attribute__((address_space(3))) v4i_vs* lv4p;  // LDS (AS3) int4*

// 16-byte global -> LDS copy. Async DMA path on CDNA5 (no VGPR staging,
// tracked by ASYNCcnt); falls back to reg-staged copy if builtin missing.
__device__ __forceinline__ void cp16(void* lds, const void* g) {
#if HAS_ASYNC_LDS
    __builtin_amdgcn_global_load_async_to_lds_b128((gv4p)g, (lv4p)lds, 0, 0);
#else
    *(uint4*)lds = *(const uint4*)g;
#endif
}

__device__ __forceinline__ void wait_cp() {
#if HAS_ASYNC_LDS
#if __has_builtin(__builtin_amdgcn_s_wait_asynccnt)
    __builtin_amdgcn_s_wait_asynccnt(0);
#else
    asm volatile("s_wait_asynccnt 0x0" ::: "memory");
#endif
#endif
}

// ---------------- scale reduction: s = mean(|w|) ----------------

__global__ void k_zero_scales(float* s) {
    if (threadIdx.x < 4) s[threadIdx.x] = 0.0f;
}

__global__ void k_abs_sum(const float* __restrict__ w, long long n,
                          float* __restrict__ out) {
    __shared__ float red[256];
    float acc = 0.0f;
    long long i      = (long long)blockIdx.x * blockDim.x + threadIdx.x;
    long long stride = (long long)gridDim.x * blockDim.x;
    for (; i < n; i += stride) acc += fabsf(w[i]);
    red[threadIdx.x] = acc;
    __syncthreads();
    for (int s = 128; s > 0; s >>= 1) {
        if (threadIdx.x < (unsigned)s) red[threadIdx.x] += red[threadIdx.x + s];
        __syncthreads();
    }
    if (threadIdx.x == 0) atomicAdd(out, red[0]);
}

// ---------------- ternary quantization to padded bf16 ----------------

// w: [HIDDEN, D_MODEL] -> t: [H_PAD, D_MODEL], pad rows zero
__global__ void k_quant_wh(const float* __restrict__ w, __bf16* __restrict__ t,
                           const float* __restrict__ ssum, float inv_count) {
    const float s    = ssum[0] * inv_count;
    const float rinv = 1.0f / s;
    long long base = ((long long)blockIdx.x * blockDim.x + threadIdx.x) * 4;
    for (int e = 0; e < 4; ++e) {
        long long idx = base + e;
        int row = (int)(idx / D_MODEL);
        float q = 0.0f;
        if (row < HIDDEN) {
            float v = w[idx] * rinv;
            q = fminf(1.0f, fmaxf(-1.0f, rintf(v)));
        }
        t[idx] = (__bf16)q;
    }
}

// w2: [D_MODEL, HIDDEN] -> t: [D_MODEL, H_PAD], pad cols zero
__global__ void k_quant_w2(const float* __restrict__ w, __bf16* __restrict__ t,
                           const float* __restrict__ ssum, float inv_count) {
    const float s    = ssum[0] * inv_count;
    const float rinv = 1.0f / s;
    long long base = ((long long)blockIdx.x * blockDim.x + threadIdx.x) * 4;
    for (int e = 0; e < 4; ++e) {
        long long idx = base + e;
        int d = (int)(idx / H_PAD);
        int h = (int)(idx % H_PAD);
        float q = 0.0f;
        if (h < HIDDEN) {
            float v = w[(long long)d * HIDDEN + h] * rinv;
            q = fminf(1.0f, fmaxf(-1.0f, rintf(v)));
        }
        t[idx] = (__bf16)q;
    }
}

__global__ void k_convert_x(const float* __restrict__ x, __bf16* __restrict__ xb) {
    long long base = ((long long)blockIdx.x * blockDim.x + threadIdx.x) * 4;
    for (int e = 0; e < 4; ++e) xb[base + e] = (__bf16)x[base + e];
}

// ---------------- GEMM1: g = silu(s1*(x@t1^T)) * (s3*(x@t3^T)) ----------------
// grid: (H_PAD/64, M_TOK/128), 256 threads = 8 waves (4 M-groups x 2 N-groups)
// Double-buffered LDS; stage k+1 filled by async global->LDS DMA while the
// 8 WMMAs of stage k execute; one barrier per K-step.

__global__ __launch_bounds__(256) void k_gemm1(
    const __bf16* __restrict__ xb, const __bf16* __restrict__ w1t,
    const __bf16* __restrict__ w3t, __bf16* __restrict__ g,
    const float* __restrict__ ssum) {
    const float sw = 1.0f / (float)((long long)HIDDEN * D_MODEL);
    const float s1 = ssum[0] * sw;
    const float s3 = ssum[1] * sw;

    __shared__ __align__(16) __bf16 sA [2][128][40];
    __shared__ __align__(16) __bf16 sB1[2][64][40];
    __shared__ __align__(16) __bf16 sB3[2][64][40];

    const int tid  = threadIdx.x;
    const int wave = tid >> 5, lane = tid & 31;
    const int wm = wave & 3, wn = wave >> 2;        // 4x2 wave grid
    const int m0 = blockIdx.y * 128, n0 = blockIdx.x * 64;

    v8f acc1[2][2] = {}; v8f acc3[2][2] = {};

    const int arow = tid >> 1, aseg = (tid & 1) * 16;  // A: 128 rows x 2 half-rows
    const int brow = tid >> 2, bseg = (tid & 3) * 8;   // B: 64 rows x 4 segs

    const __bf16* gA  = xb  + (long long)(m0 + arow) * D_MODEL + aseg;
    const __bf16* gB1 = w1t + (long long)(n0 + brow) * D_MODEL + bseg;
    const __bf16* gB3 = w3t + (long long)(n0 + brow) * D_MODEL + bseg;

    auto load_stage = [&](int buf, int k0) {
        cp16(&sA[buf][arow][aseg],     gA + k0);
        cp16(&sA[buf][arow][aseg + 8], gA + k0 + 8);
        cp16(&sB1[buf][brow][bseg],    gB1 + k0);
        cp16(&sB3[buf][brow][bseg],    gB3 + k0);
    };

    load_stage(0, 0);
    wait_cp();
    __syncthreads();

    const int r = lane & 15, kb = (lane >> 4) << 3;  // ISA 16-bit A 16x32 layout
    const int KT = D_MODEL / 32;

    for (int kt = 0; kt < KT; ++kt) {
        const int cur = kt & 1;
        if (kt + 1 < KT) load_stage(cur ^ 1, (kt + 1) * 32);

        FragBF a[2], b1[2], b3[2];
        #pragma unroll
        for (int i = 0; i < 2; ++i) {
            const __bf16* p = &sA[cur][wm * 32 + i * 16 + r][0];
            a[i].h[0] = *(const v8bf*)(p + kb);
            a[i].h[1] = *(const v8bf*)(p + kb + 16);
        }
        #pragma unroll
        for (int j = 0; j < 2; ++j) {
            const __bf16* p1 = &sB1[cur][wn * 32 + j * 16 + r][0];
            b1[j].h[0] = *(const v8bf*)(p1 + kb);
            b1[j].h[1] = *(const v8bf*)(p1 + kb + 16);
            const __bf16* p3 = &sB3[cur][wn * 32 + j * 16 + r][0];
            b3[j].h[0] = *(const v8bf*)(p3 + kb);
            b3[j].h[1] = *(const v8bf*)(p3 + kb + 16);
        }
        #pragma unroll
        for (int i = 0; i < 2; ++i)
            #pragma unroll
            for (int j = 0; j < 2; ++j) {
                acc1[i][j] = __builtin_amdgcn_wmma_f32_16x16x32_bf16(
                    false, a[i].v, false, b1[j].v, (short)0, acc1[i][j], false, false);
                acc3[i][j] = __builtin_amdgcn_wmma_f32_16x16x32_bf16(
                    false, a[i].v, false, b3[j].v, (short)0, acc3[i][j], false, false);
            }
        wait_cp();
        __syncthreads();
    }

    // epilogue: C/D 16x16 f32 layout -> silu gate -> bf16 store
    const int hi = lane >> 4;
    #pragma unroll
    for (int i = 0; i < 2; ++i)
        #pragma unroll
        for (int j = 0; j < 2; ++j)
            #pragma unroll
            for (int e = 0; e < 8; ++e) {
                int mrow = m0 + wm * 32 + i * 16 + e + (hi << 3);
                int ncol = n0 + wn * 32 + j * 16 + r;
                float h1 = s1 * acc1[i][j][e];
                float h3 = s3 * acc3[i][j][e];
                float gv = (h1 / (1.0f + __expf(-h1))) * h3;
                g[(long long)mrow * H_PAD + ncol] = (__bf16)gv;
            }
}

// ---------------- GEMM2: out = s2 * (g @ t2^T), fp32 ----------------
// grid: (D_MODEL/64, M_TOK/128)

__global__ __launch_bounds__(256) void k_gemm2(
    const __bf16* __restrict__ g, const __bf16* __restrict__ w2t,
    float* __restrict__ out, const float* __restrict__ ssum) {
    const float s2 = ssum[2] * (1.0f / (float)((long long)HIDDEN * D_MODEL));

    __shared__ __align__(16) __bf16 sA[2][128][40];
    __shared__ __align__(16) __bf16 sB[2][64][40];

    const int tid  = threadIdx.x;
    const int wave = tid >> 5, lane = tid & 31;
    const int wm = wave & 3, wn = wave >> 2;
    const int m0 = blockIdx.y * 128, n0 = blockIdx.x * 64;

    v8f acc[2][2] = {};

    const int arow = tid >> 1, aseg = (tid & 1) * 16;
    const int brow = tid >> 2, bseg = (tid & 3) * 8;

    const __bf16* gA = g   + (long long)(m0 + arow) * H_PAD + aseg;
    const __bf16* gB = w2t + (long long)(n0 + brow) * H_PAD + bseg;

    auto load_stage = [&](int buf, int k0) {
        cp16(&sA[buf][arow][aseg],     gA + k0);
        cp16(&sA[buf][arow][aseg + 8], gA + k0 + 8);
        cp16(&sB[buf][brow][bseg],     gB + k0);
    };

    load_stage(0, 0);
    wait_cp();
    __syncthreads();

    const int r = lane & 15, kb = (lane >> 4) << 3;
    const int KT = H_PAD / 32;

    for (int kt = 0; kt < KT; ++kt) {
        const int cur = kt & 1;
        if (kt + 1 < KT) load_stage(cur ^ 1, (kt + 1) * 32);

        FragBF a[2], b[2];
        #pragma unroll
        for (int i = 0; i < 2; ++i) {
            const __bf16* p = &sA[cur][wm * 32 + i * 16 + r][0];
            a[i].h[0] = *(const v8bf*)(p + kb);
            a[i].h[1] = *(const v8bf*)(p + kb + 16);
        }
        #pragma unroll
        for (int j = 0; j < 2; ++j) {
            const __bf16* p = &sB[cur][wn * 32 + j * 16 + r][0];
            b[j].h[0] = *(const v8bf*)(p + kb);
            b[j].h[1] = *(const v8bf*)(p + kb + 16);
        }
        #pragma unroll
        for (int i = 0; i < 2; ++i)
            #pragma unroll
            for (int j = 0; j < 2; ++j)
                acc[i][j] = __builtin_amdgcn_wmma_f32_16x16x32_bf16(
                    false, a[i].v, false, b[j].v, (short)0, acc[i][j], false, false);
        wait_cp();
        __syncthreads();
    }

    const int hi = lane >> 4;
    #pragma unroll
    for (int i = 0; i < 2; ++i)
        #pragma unroll
        for (int j = 0; j < 2; ++j)
            #pragma unroll
            for (int e = 0; e < 8; ++e) {
                int mrow = m0 + wm * 32 + i * 16 + e + (hi << 3);
                int ncol = n0 + wn * 32 + j * 16 + r;
                out[(long long)mrow * D_MODEL + ncol] = s2 * acc[i][j][e];
            }
}

// ---------------- launcher ----------------

extern "C" void kernel_launch(void* const* d_in, const int* in_sizes, int n_in,
                              void* d_out, int out_size, void* d_ws, size_t ws_size,
                              hipStream_t stream) {
    const float* x  = (const float*)d_in[0];
    const float* w1 = (const float*)d_in[1];
    const float* w3 = (const float*)d_in[2];
    const float* w2 = (const float*)d_in[3];
    float* out = (float*)d_out;

    char* ws = (char*)d_ws;
    float* scales = (float*)ws;
    size_t off = 256;
    __bf16* xb  = (__bf16*)(ws + off); off += (size_t)M_TOK * D_MODEL * 2;
    __bf16* w1t = (__bf16*)(ws + off); off += (size_t)H_PAD * D_MODEL * 2;
    __bf16* w3t = (__bf16*)(ws + off); off += (size_t)H_PAD * D_MODEL * 2;
    __bf16* w2t = (__bf16*)(ws + off); off += (size_t)D_MODEL * H_PAD * 2;
    __bf16* gbuf = (__bf16*)(ws + off);

    k_zero_scales<<<1, 64, 0, stream>>>(scales);

    const long long nw = (long long)HIDDEN * D_MODEL;
    k_abs_sum<<<1024, 256, 0, stream>>>(w1, nw, scales + 0);
    k_abs_sum<<<1024, 256, 0, stream>>>(w3, nw, scales + 1);
    k_abs_sum<<<1024, 256, 0, stream>>>(w2, nw, scales + 2);

    const float invc = 1.0f / (float)nw;
    {
        long long n = (long long)H_PAD * D_MODEL;  // divisible by 1024
        int blocks = (int)(n / (256 * 4));
        k_quant_wh<<<blocks, 256, 0, stream>>>(w1, w1t, scales + 0, invc);
        k_quant_wh<<<blocks, 256, 0, stream>>>(w3, w3t, scales + 1, invc);
        k_quant_w2<<<blocks, 256, 0, stream>>>(w2, w2t, scales + 2, invc);
    }
    {
        long long n = (long long)M_TOK * D_MODEL;  // divisible by 1024
        k_convert_x<<<(int)(n / (256 * 4)), 256, 0, stream>>>(x, xb);
    }

    dim3 g1(H_PAD / 64, M_TOK / 128);
    k_gemm1<<<g1, 256, 0, stream>>>(xb, w1t, w3t, gbuf, scales);

    dim3 g2(D_MODEL / 64, M_TOK / 128);
    k_gemm2<<<g2, 256, 0, stream>>>(gbuf, w2t, out, scales);
}
